// Decoder_70231305224764
// MI455X (gfx1250) — matrix operand order repaired and checked by reading, not compiled, not run on hardware
//
#include <hip/hip_runtime.h>
#include <hip/hip_bf16.h>

// ---------------------------------------------------------------------------
// GRU decoder on gfx1250: bf16 WMMA (v_wmma_f32_16x16x32_bf16), f32 accumulate.
// Weight slices per block are DMA'd into LDS by the Tensor Data Mover
// (tensor_load_to_lds, TENSORcnt), overlapped with the gi K-loop; B-fragments
// then come from LDS (ds_load_b128), A-fragments from global (L2-resident h).
// Sizes fixed by reference: B=256, LATENT=128, H=1024, O=256, T=512.
// ---------------------------------------------------------------------------

typedef __attribute__((ext_vector_type(16))) __bf16 v16bf;
typedef __attribute__((ext_vector_type(8)))  float  v8f;
typedef __attribute__((ext_vector_type(4)))  unsigned int v4u;
typedef __attribute__((ext_vector_type(8)))  int    v8i;
typedef __attribute__((ext_vector_type(4)))  int    v4i;

#define BATCH 256
#define LATENT 128
#define HID   1024
#define OUTD  256
#define TLEN  512

#if __has_builtin(__builtin_amdgcn_tensor_load_to_lds) && \
    __has_builtin(__builtin_amdgcn_s_wait_tensorcnt)
#define USE_TDM 1
#endif

__device__ __forceinline__ v8f wmma_bf16(v16bf a, v16bf b, v8f c) {
  // 8 args: (neg_a, A, neg_b, B, c_mod, C, reuse_a, reuse_b)
  return __builtin_amdgcn_wmma_f32_16x16x32_bf16(false, a, false, b, (short)0, c,
                                                 false, false);
}

// A fragment (16x32 bf16), row-major source with leading dim `ld`.
// ISA 7.12.2 16-bit A layout: lanes 0-15 (row M=lane) hold K {k0..+7, k0+16..+23},
// lanes 16-31 (row M=lane-16) hold K {k0+8..+15, k0+24..+31}.
__device__ __forceinline__ v16bf load_a_frag(const __bf16* __restrict__ A, int ld,
                                             int row_base, int k0, int lane) {
  int r  = lane & 15;
  int hl = lane >> 4;
  const __bf16* p = A + (size_t)(row_base + r) * ld + k0;
  union { v16bf v; float4 f[2]; } u;
  u.f[0] = *(const float4*)(p + 8 * hl);
  u.f[1] = *(const float4*)(p + 16 + 8 * hl);
  return u.v;
}

// B fragment (32x16 bf16) for D = A*B with B[k][n] = W[col_base+n][k],
// W row-major, leading dim `ld` (GEMM against W^T). Lanes 0-15: col n=lane,
// K {k0..+7, k0+16..+23}; lanes 16-31: K {k0+8..+15, k0+24..+31} (matches A).
// Works for both global and LDS-resident W.
__device__ __forceinline__ v16bf load_b_frag(const __bf16* W, int ld,
                                             int col_base, int k0, int lane) {
  int c  = lane & 15;
  int hl = lane >> 4;
  const __bf16* p = W + (size_t)(col_base + c) * ld + k0 + 8 * hl;
  union { v16bf v; float4 f[2]; } u;
  u.f[0] = *(const float4*)(p);
  u.f[1] = *(const float4*)(p + 16);
  return u.v;
}

__device__ __forceinline__ float sigmoidf_fast(float x) {
  return 1.0f / (1.0f + __expf(-x));
}

#ifdef USE_TDM
// Issue one TDM 2D tile load: ysize row-blocks of xsize contiguous bf16
// elements, row-block stride `stride0` (elements), into LDS byte offset
// `lds_off` (contiguous). D# bitfields per CDNA5 ISA §8 (data_size=2B,
// count=1, type=2). Groups 2/3 zero (2D tensor). Wave-uniform operands.
__device__ __forceinline__ void tdm_load_2d(const void* gbase, unsigned lds_off,
                                            unsigned xsize, unsigned ysize,
                                            unsigned long long stride0) {
  unsigned long long ga = (unsigned long long)(size_t)gbase;
  v4u g0;
  g0[0] = 1u;                                        // count=1, user mode
  g0[1] = lds_off;                                   // lds_addr [63:32]
  g0[2] = (unsigned)(ga & 0xFFFFFFFFu);              // global_addr [95:64]
  g0[3] = (unsigned)((ga >> 32) & 0x01FFFFFFu)       // global_addr [120:96]
        | (2u << 30);                                // type=2 ("image")
  v8i g1;
  g1[0] = (int)(1u << 16);                           // data_size=1 -> 2 bytes
  g1[1] = (int)((xsize & 0xFFFFu) << 16);            // tensor_dim0[15:0]
  g1[2] = (int)(((xsize >> 16) & 0xFFFFu)            // tensor_dim0[31:16]
        | ((ysize & 0xFFFFu) << 16));                // tensor_dim1[15:0]
  g1[3] = (int)((xsize & 0xFFFFu) << 16);            // tile_dim0 (16-bit)
  g1[4] = (int)(ysize & 0xFFFFu);                    // tile_dim1, tile_dim2=0
  g1[5] = (int)(unsigned)(stride0 & 0xFFFFFFFFull);  // tensor_dim0_stride lo
  g1[6] = (int)(unsigned)((stride0 >> 32) & 0xFFFFull); // stride hi
  g1[7] = 0;
  v4i zz4 = {0, 0, 0, 0};
  v8i zz8 = {0, 0, 0, 0, 0, 0, 0, 0};
  // 6-arg toolchain variant: (g0, g1, g2, g3, g4, cpol)
  __builtin_amdgcn_tensor_load_to_lds(g0, g1, zz4, zz4, zz8, 0);
}
#endif

// ---------------------------------------------------------------------------
// Fused GRU gate kernel. Grid: 64 blocks x 256 threads (8 waves).
// Block owns one 16-col slice n0 = blockIdx*16; its 8 waves cover all 256 rows
// (wave tile 32Mx16N). Weight slices (W_hh: 3x16x1024, W_ih: 3x16x256, bf16)
// live in LDS. Dynamic LDS: 122880 bytes.
// ---------------------------------------------------------------------------
__global__ void __launch_bounds__(256) gru_gates_kernel(
    const __bf16* __restrict__ h_bf, const float* __restrict__ h_f32,
    const __bf16* __restrict__ x_bf,
    const __bf16* __restrict__ Wih_b, const __bf16* __restrict__ Whh_b,
    const float* __restrict__ b_ih, const float* __restrict__ b_hh,
    float* __restrict__ h_f32_out, __bf16* __restrict__ h_bf_out)
{
  extern __shared__ char lds_raw[];
  __bf16* Whh_s = (__bf16*)lds_raw;              // [48][1024] (gate*16+c rows)
  __bf16* Wih_s = Whh_s + 48 * HID;              // [48][256]

  const int tid  = (int)threadIdx.x;
  const int lane = tid & 31;
  const int tileM = tid >> 5;                    // 8 waves -> 8 M tiles of 32
  const int m0 = tileM * 32;
  const int n0 = (int)blockIdx.x * 16;           // 64 blocks -> 1024 cols

#ifdef USE_TDM
  // TDM: W_ih slice first (needed by gi loop), then W_hh slice; TDM ops from
  // one wave are in-order, so tensorcnt<=1 means the W_ih tile has landed.
  if (tid < 32) {
    tdm_load_2d(Wih_b + (size_t)n0 * OUTD, (unsigned)(48 * HID * 2),
                16 * OUTD, 3, (unsigned long long)HID * OUTD);
    tdm_load_2d(Whh_b + (size_t)n0 * HID, 0u,
                16 * HID, 3, (unsigned long long)HID * HID);
    __builtin_amdgcn_s_wait_tensorcnt(1);
  }
  __syncthreads();
#else
  // Fallback: cooperative VGPR staging.
  {
    uint4* dst = (uint4*)Whh_s;                  // 48 rows x 128 uint4
    for (int i = tid; i < 48 * 128; i += 256) {
      int row = i >> 7, q = i & 127;
      int g = row >> 4, c = row & 15;
      dst[i] = *((const uint4*)(Whh_b + (size_t)(g * HID + n0 + c) * HID) + q);
    }
    uint4* dst2 = (uint4*)Wih_s;                 // 48 rows x 32 uint4
    for (int i = tid; i < 48 * 32; i += 256) {
      int row = i >> 5, q = i & 31;
      int g = row >> 4, c = row & 15;
      dst2[i] = *((const uint4*)(Wih_b + (size_t)(g * HID + n0 + c) * OUTD) + q);
    }
  }
  __syncthreads();
#endif

  v8f acc_r0{}, acc_r1{}, acc_z0{}, acc_z1{};
  v8f acc_in0{}, acc_in1{}, acc_hn0{}, acc_hn1{};   // i_n and h_n kept separate

  // ---- gi: x (B x O) against W_ih slice (LDS); overlaps W_hh TDM stream ----
  for (int k = 0; k < OUTD; k += 32) {
    v16bf a0 = load_a_frag(x_bf, OUTD, m0,      k, lane);
    v16bf a1 = load_a_frag(x_bf, OUTD, m0 + 16, k, lane);
    v16bf br = load_b_frag(Wih_s, OUTD,  0, k, lane);
    v16bf bz = load_b_frag(Wih_s, OUTD, 16, k, lane);
    v16bf bn = load_b_frag(Wih_s, OUTD, 32, k, lane);
    acc_r0  = wmma_bf16(a0, br, acc_r0);  acc_r1  = wmma_bf16(a1, br, acc_r1);
    acc_z0  = wmma_bf16(a0, bz, acc_z0);  acc_z1  = wmma_bf16(a1, bz, acc_z1);
    acc_in0 = wmma_bf16(a0, bn, acc_in0); acc_in1 = wmma_bf16(a1, bn, acc_in1);
  }

#ifdef USE_TDM
  if (tid < 32) __builtin_amdgcn_s_wait_tensorcnt(0);
  __syncthreads();
#endif

  // ---- gh: h (B x H) against W_hh slice (LDS) ----
  for (int k = 0; k < HID; k += 32) {
    v16bf a0 = load_a_frag(h_bf, HID, m0,      k, lane);
    v16bf a1 = load_a_frag(h_bf, HID, m0 + 16, k, lane);
    v16bf br = load_b_frag(Whh_s, HID,  0, k, lane);
    v16bf bz = load_b_frag(Whh_s, HID, 16, k, lane);
    v16bf bn = load_b_frag(Whh_s, HID, 32, k, lane);
    acc_r0  = wmma_bf16(a0, br, acc_r0);  acc_r1  = wmma_bf16(a1, br, acc_r1);
    acc_z0  = wmma_bf16(a0, bz, acc_z0);  acc_z1  = wmma_bf16(a1, bz, acc_z1);
    acc_hn0 = wmma_bf16(a0, bn, acc_hn0); acc_hn1 = wmma_bf16(a1, bn, acc_hn1);
  }

  // ---- gate math + state update ----
  int nCol = n0 + (lane & 15);
  int hl   = lane >> 4;
  float bir = b_ih[nCol]           + b_hh[nCol];
  float biz = b_ih[HID + nCol]     + b_hh[HID + nCol];
  float bin = b_ih[2 * HID + nCol];
  float bhn = b_hh[2 * HID + nCol];

  #pragma unroll
  for (int mt = 0; mt < 2; ++mt) {
    v8f& ar  = mt ? acc_r1  : acc_r0;
    v8f& az  = mt ? acc_z1  : acc_z0;
    v8f& ain = mt ? acc_in1 : acc_in0;
    v8f& ahn = mt ? acc_hn1 : acc_hn0;
    #pragma unroll
    for (int v = 0; v < 8; ++v) {
      // C/D layout: VGPR v -> M = v (lanes 0-15) or 8+v (lanes 16-31)
      int m = m0 + mt * 16 + hl * 8 + v;
      float rg = sigmoidf_fast(ar[v] + bir);
      float ug = sigmoidf_fast(az[v] + biz);
      float nn = tanhf((ain[v] + bin) + rg * (ahn[v] + bhn));
      float ho = h_f32[(size_t)m * HID + nCol];
      float hv = (1.0f - ug) * nn + ug * ho;
      h_f32_out[(size_t)m * HID + nCol] = hv;
      h_bf_out [(size_t)m * HID + nCol] = (__bf16)hv;
    }
  }
}

// ---------------------------------------------------------------------------
// Output projection: out = h_new @ W_out^T + b_out; writes d_out[:,t,:] and
// the bf16 feedback input x. Grid: 16 blocks x 256 threads; block shares one
// 16-col slice of W_out in LDS (32 KB dynamic), TDM-loaded.
// ---------------------------------------------------------------------------
__global__ void __launch_bounds__(256) gru_out_kernel(
    const __bf16* __restrict__ h_bf, const __bf16* __restrict__ Wout_b,
    const float* __restrict__ b_out,
    float* __restrict__ out, int t, __bf16* __restrict__ x_bf)
{
  extern __shared__ char lds_raw[];
  __bf16* Wout_s = (__bf16*)lds_raw;             // [16][1024]

  const int tid  = (int)threadIdx.x;
  const int lane = tid & 31;
  const int tileM = tid >> 5;
  const int m0 = tileM * 32;
  const int o0 = (int)blockIdx.x * 16;           // 16 blocks -> 256 cols

#ifdef USE_TDM
  if (tid < 32) {
    tdm_load_2d(Wout_b + (size_t)o0 * HID, 0u, 16 * HID, 1,
                (unsigned long long)16 * HID);
    __builtin_amdgcn_s_wait_tensorcnt(0);
  }
  __syncthreads();
#else
  {
    uint4* dst = (uint4*)Wout_s;                 // 16 rows x 128 uint4
    for (int i = tid; i < 16 * 128; i += 256) {
      int row = i >> 7, q = i & 127;
      dst[i] = *((const uint4*)(Wout_b + (size_t)(o0 + row) * HID) + q);
    }
  }
  __syncthreads();
#endif

  v8f acc0{}, acc1{};
  for (int k = 0; k < HID; k += 32) {
    v16bf a0 = load_a_frag(h_bf, HID, m0,      k, lane);
    v16bf a1 = load_a_frag(h_bf, HID, m0 + 16, k, lane);
    v16bf bb = load_b_frag(Wout_s, HID, 0, k, lane);
    acc0 = wmma_bf16(a0, bb, acc0);
    acc1 = wmma_bf16(a1, bb, acc1);
  }

  int o  = o0 + (lane & 15);
  int hl = lane >> 4;
  float bo = b_out[o];
  #pragma unroll
  for (int mt = 0; mt < 2; ++mt) {
    v8f& a = mt ? acc1 : acc0;
    #pragma unroll
    for (int v = 0; v < 8; ++v) {
      int m = m0 + mt * 16 + hl * 8 + v;
      float val = a[v] + bo;
      out[(size_t)m * TLEN * OUTD + (size_t)t * OUTD + o] = val;
      x_bf[(size_t)m * OUTD + o] = (__bf16)val;
    }
  }
}

// ---------------------------------------------------------------------------
// Init: h0 = tanh(z @ W_lh^T + b_lh), plain f32 (one-time 67 MFLOP).
// ---------------------------------------------------------------------------
__global__ void __launch_bounds__(256) gru_h0_kernel(
    const float* __restrict__ z, const float* __restrict__ Wlh,
    const float* __restrict__ blh,
    float* __restrict__ h_f32, __bf16* __restrict__ h_bf)
{
  int idx = (int)(blockIdx.x * blockDim.x + threadIdx.x);   // 256*1024
  int b = idx >> 10;
  int j = idx & 1023;
  const float* zr = z   + (size_t)b * LATENT;
  const float* wr = Wlh + (size_t)j * LATENT;
  float s = blh[j];
  #pragma unroll 4
  for (int k = 0; k < LATENT; ++k) s = fmaf(zr[k], wr[k], s);
  float h = tanhf(s);
  h_f32[idx] = h;
  h_bf [idx] = (__bf16)h;
}

__global__ void cvt_f32_to_bf16(const float* __restrict__ src,
                                __bf16* __restrict__ dst, int n) {
  for (int i = (int)(blockIdx.x * blockDim.x + threadIdx.x); i < n;
       i += (int)(gridDim.x * blockDim.x))
    dst[i] = (__bf16)src[i];
}

__global__ void zero_bf16_kernel(__bf16* __restrict__ dst, int n) {
  for (int i = (int)(blockIdx.x * blockDim.x + threadIdx.x); i < n;
       i += (int)(gridDim.x * blockDim.x))
    dst[i] = (__bf16)0.0f;
}

// ---------------------------------------------------------------------------
extern "C" void kernel_launch(void* const* d_in, const int* in_sizes, int n_in,
                              void* d_out, int out_size, void* d_ws, size_t ws_size,
                              hipStream_t stream) {
  const float* z     = (const float*)d_in[0];
  const float* W_lh  = (const float*)d_in[1];
  const float* b_lh  = (const float*)d_in[2];
  const float* W_ih  = (const float*)d_in[3];
  const float* b_ih  = (const float*)d_in[4];
  const float* W_hh  = (const float*)d_in[5];
  const float* b_hh  = (const float*)d_in[6];
  const float* W_out = (const float*)d_in[7];
  const float* b_out = (const float*)d_in[8];
  float* out = (float*)d_out;

  // Workspace carve-up (~11.7 MB total)
  char* ws = (char*)d_ws;
  size_t off = 0;
  auto carve = [&](size_t bytes) -> void* {
    void* p = ws + off;
    off += (bytes + 255) & ~(size_t)255;
    return p;
  };
  __bf16* Wih_b  = (__bf16*)carve((size_t)3 * HID * OUTD * 2);
  __bf16* Whh_b  = (__bf16*)carve((size_t)3 * HID * HID  * 2);
  __bf16* Wout_b = (__bf16*)carve((size_t)OUTD * HID * 2);
  float*  h32[2];
  __bf16* hbf[2];
  h32[0] = (float*) carve((size_t)BATCH * HID * 4);
  h32[1] = (float*) carve((size_t)BATCH * HID * 4);
  hbf[0] = (__bf16*)carve((size_t)BATCH * HID * 2);
  hbf[1] = (__bf16*)carve((size_t)BATCH * HID * 2);
  __bf16* x_bf = (__bf16*)carve((size_t)BATCH * OUTD * 2);

  // One-time weight conversion to bf16 (stays L2-resident thereafter)
  cvt_f32_to_bf16<<<1024, 256, 0, stream>>>(W_ih,  Wih_b,  3 * HID * OUTD);
  cvt_f32_to_bf16<<<1024, 256, 0, stream>>>(W_hh,  Whh_b,  3 * HID * HID);
  cvt_f32_to_bf16<<<1024, 256, 0, stream>>>(W_out, Wout_b, OUTD * HID);

  // h0 and zero first input
  gru_h0_kernel<<<(BATCH * HID) / 256, 256, 0, stream>>>(z, W_lh, b_lh, h32[0], hbf[0]);
  zero_bf16_kernel<<<64, 256, 0, stream>>>(x_bf, BATCH * OUTD);

  const size_t lds_gates = (size_t)(48 * HID + 48 * OUTD) * sizeof(__bf16); // 122880
  const size_t lds_out   = (size_t)(16 * HID) * sizeof(__bf16);             // 32768

  // Sequential recurrence: 512 steps, 2 kernels each, ping-pong h buffers.
  int cur = 0;
  for (int t = 0; t < TLEN; ++t) {
    int nxt = cur ^ 1;
    gru_gates_kernel<<<64, 256, lds_gates, stream>>>(hbf[cur], h32[cur], x_bf,
                                                     Wih_b, Whh_b, b_ih, b_hh,
                                                     h32[nxt], hbf[nxt]);
    gru_out_kernel<<<16, 256, lds_out, stream>>>(hbf[nxt], Wout_b, b_out, out, t, x_bf);
    cur = nxt;
  }
}
